// PCT_semseg_2662879723818
// MI455X (gfx1250) — matrix operand order repaired and checked by reading, not compiled, not run on hardware
//
#include <hip/hip_runtime.h>
#include <hip/hip_bf16.h>

// ---------------------------------------------------------------------------
// PCT semantic-segmentation forward for MI455X (gfx1250, wave32, WMMA).
// All heavy matmuls go through a bf16 WMMA GEMM (V_WMMA_F32_16X16X32_BF16)
// with fp32 accumulation and fused BN/bias/activation/col-renorm epilogue.
// v4: 6-arg TDM builtins (clang-23 arity) for the channel-concat block copies.
// ---------------------------------------------------------------------------

typedef __bf16 bf16;
typedef __attribute__((ext_vector_type(16))) __bf16 v16bf;
typedef __attribute__((ext_vector_type(8)))  __bf16 v8bf;
typedef __attribute__((ext_vector_type(4)))  __bf16 v4bf;
typedef __attribute__((ext_vector_type(8)))  float  v8f;
typedef __attribute__((ext_vector_type(4)))  unsigned int v4u;
typedef __attribute__((ext_vector_type(8)))  int v8i;
typedef __attribute__((ext_vector_type(4)))  int v4i;

#define NPTS  2048
#define BATCH 4
#define KNN   20
#define NSEG  13
#define BKPAD 48   // LDS row pitch in bf16 (32 data + 16 pad, 96B rows)

#if __has_builtin(__builtin_amdgcn_tensor_load_to_lds) && \
    __has_builtin(__builtin_amdgcn_tensor_store_from_lds) && \
    __has_builtin(__builtin_amdgcn_s_wait_tensorcnt)
#define HAVE_TDM 1
#else
#define HAVE_TDM 0
#endif

// ---------------------------------------------------------------------------
// Fold inference BN (+ optional conv bias) into per-channel scale/shift.
// ---------------------------------------------------------------------------
__global__ void k_fold_bn(const float* g, const float* bt, const float* m,
                          const float* v, const float* bias,
                          float* sc, float* sh, int C) {
  int i = blockIdx.x * 256 + threadIdx.x;
  if (i >= C) return;
  float s = g[i] * rsqrtf(v[i] + 1e-5f);
  float b0 = bias ? bias[i] : 0.f;
  sc[i] = s;
  sh[i] = s * (b0 - m[i]) + bt[i];
}

// ---------------------------------------------------------------------------
// bf16 WMMA GEMM: D[m][n] = act(scale[m]*sum_k A[m][k]*B[k][n] + shift[m])
//                 optionally / (1e-6 + coldiv[n])
// Requirements: M%32==0, K%32==0, Ncols%128==0 (guaranteed by callers).
// Block: 256 threads = 8 waves. Tile 32(M) x 128(N), K stepped by 32.
// Waves: 2(M) x 4(N); each wave computes two 16x16 tiles sharing one A frag.
// act: 0=none, 1=relu, 2=leaky-relu(0.2)
// ---------------------------------------------------------------------------
__global__ __launch_bounds__(256)
void k_gemm_wmma(int M, int K, int Ncols,
                 const float* A, long Ar, long Ac, long Abat,
                 const float* B, long Br, long Bc, long Bbat,
                 float* D, long ldD, long Dbat,
                 const float* scale, const float* shift,
                 const float* coldiv, long cdBat, int act) {
  __shared__ bf16 As[2][32 * BKPAD];    // [m][k] (k contiguous, padded)
  __shared__ bf16 Bs[2][128 * BKPAD];   // [n][k] (k-major staging, padded)

  int b = blockIdx.z;
  A += (long)b * Abat;
  B += (long)b * Bbat;
  D += (long)b * Dbat;

  int m0 = blockIdx.y * 32;
  int n0 = blockIdx.x * 128;
  int tid  = threadIdx.x;
  int lane = tid & 31;
  int wave = tid >> 5;
  int mw = wave >> 2;   // 0..1
  int nw = wave & 3;    // 0..3
  int half = lane >> 4;
  int l15  = lane & 15;

  const bool fastA = (Ac == 1) && ((Ar & 3) == 0);
  const bool fastB = (Bc == 1) && ((Br & 3) == 0);

  // Stage one 32-wide K slice (A: 32x32, B: 32x128) into LDS buffer `buf`,
  // converting fp32 -> bf16. Loads are issued in a batch (kept in registers)
  // before the LDS stores so multiple b128 loads stay in flight.
  auto stage = [&](int k0, int buf) {
    bf16* as = &As[buf][0];
    bf16* bs = &Bs[buf][0];
    if (fastA) {
      int m  = tid >> 3;
      int k4 = (tid & 7) * 4;
      float4 v = *(const float4*)&A[(long)(m0 + m) * Ar + (k0 + k4)];
      v4bf pk;
      pk[0] = (bf16)v.x; pk[1] = (bf16)v.y; pk[2] = (bf16)v.z; pk[3] = (bf16)v.w;
      *(v4bf*)&as[m * BKPAD + k4] = pk;
    } else {
      float tmp[4];
#pragma unroll
      for (int it = 0; it < 4; ++it) {
        int i = tid + it * 256;
        int m = i & 31, k = i >> 5;
        tmp[it] = A[(long)(m0 + m) * Ar + (long)(k0 + k) * Ac];
      }
#pragma unroll
      for (int it = 0; it < 4; ++it) {
        int i = tid + it * 256;
        int m = i & 31, k = i >> 5;
        as[m * BKPAD + k] = (bf16)tmp[it];
      }
    }
    if (fastB) {
      float4 v[4];
#pragma unroll
      for (int it = 0; it < 4; ++it) {
        int g  = tid + it * 256;           // 1024 groups of 4 columns
        int k  = g >> 5;
        int n4 = (g & 31) * 4;
        v[it] = *(const float4*)&B[(long)(k0 + k) * Br + (n0 + n4)];
      }
#pragma unroll
      for (int it = 0; it < 4; ++it) {
        int g  = tid + it * 256;
        int k  = g >> 5;
        int n4 = (g & 31) * 4;
        bs[(n4 + 0) * BKPAD + k] = (bf16)v[it].x;
        bs[(n4 + 1) * BKPAD + k] = (bf16)v[it].y;
        bs[(n4 + 2) * BKPAD + k] = (bf16)v[it].z;
        bs[(n4 + 3) * BKPAD + k] = (bf16)v[it].w;
      }
    } else {
      float tmp[16];
#pragma unroll
      for (int it = 0; it < 16; ++it) {
        int i = tid + it * 256;
        int k = i >> 7, n = i & 127;
        tmp[it] = B[(long)(k0 + k) * Br + (long)(n0 + n) * Bc];
      }
#pragma unroll
      for (int it = 0; it < 16; ++it) {
        int i = tid + it * 256;
        int k = i >> 7, n = i & 127;
        bs[n * BKPAD + k] = (bf16)tmp[it];
      }
    }
  };

  v8f acc0 = {0.f, 0.f, 0.f, 0.f, 0.f, 0.f, 0.f, 0.f};
  v8f acc1 = {0.f, 0.f, 0.f, 0.f, 0.f, 0.f, 0.f, 0.f};

  stage(0, 0);
  __syncthreads();
  int cur = 0;
  for (int k0 = 0; k0 < K; k0 += 32) {
    if (k0 + 32 < K) stage(k0 + 32, cur ^ 1);   // overlap next slice with WMMA
    if (k0 + 64 < K) {
      __builtin_prefetch(&A[(long)m0 * Ar + (long)(k0 + 64) * Ac], 0, 1);
      __builtin_prefetch(&B[(long)(k0 + 64) * Br + (long)n0 * Bc], 0, 1);
    }
    const bf16* as = &As[cur][0];
    const bf16* bs = &Bs[cur][0];
    // A fragment (16x32): lanes 0-15 rows M; VGPRs 0-3: K=(half*16)+0..7,
    // VGPRs 4-7: K=(half*16)+8..15. Two aligned 16B LDS reads.
    v8bf alo = *(const v8bf*)&as[(mw * 16 + l15) * BKPAD + half * 16];
    v8bf ahi = *(const v8bf*)&as[(mw * 16 + l15) * BKPAD + half * 16 + 8];
    v16bf a;
#pragma unroll
    for (int j = 0; j < 8; ++j) { a[j] = alo[j]; a[8 + j] = ahi[j]; }
    // Two N-subtiles per wave, shared A fragment.
    {
      v8bf blo = *(const v8bf*)&bs[(nw * 32 + l15) * BKPAD + half * 16];
      v8bf bhi = *(const v8bf*)&bs[(nw * 32 + l15) * BKPAD + half * 16 + 8];
      v16bf bb;
#pragma unroll
      for (int j = 0; j < 8; ++j) { bb[j] = blo[j]; bb[8 + j] = bhi[j]; }
      acc0 = __builtin_amdgcn_wmma_f32_16x16x32_bf16(
          false, a, false, bb, (short)0, acc0, false, false);
    }
    {
      v8bf blo = *(const v8bf*)&bs[(nw * 32 + 16 + l15) * BKPAD + half * 16];
      v8bf bhi = *(const v8bf*)&bs[(nw * 32 + 16 + l15) * BKPAD + half * 16 + 8];
      v16bf bb;
#pragma unroll
      for (int j = 0; j < 8; ++j) { bb[j] = blo[j]; bb[8 + j] = bhi[j]; }
      acc1 = __builtin_amdgcn_wmma_f32_16x16x32_bf16(
          false, a, false, bb, (short)0, acc1, false, false);
    }
    __syncthreads();
    cur ^= 1;
  }

  // Epilogue. D layout: VGPR r -> row m = r + 8*(lane>>4), col n = lane&15.
#pragma unroll
  for (int t = 0; t < 2; ++t) {
    v8f acc = t ? acc1 : acc0;
    int nn = n0 + nw * 32 + t * 16 + l15;
#pragma unroll
    for (int r = 0; r < 8; ++r) {
      int mm = m0 + mw * 16 + r + 8 * half;
      float vv = acc[r];
      if (scale) vv = vv * scale[mm] + (shift ? shift[mm] : 0.f);
      else if (shift) vv += shift[mm];
      if (act == 1)      vv = vv > 0.f ? vv : 0.f;
      else if (act == 2) vv = vv > 0.f ? vv : 0.2f * vv;
      if (coldiv) vv /= (1e-6f + coldiv[(long)b * cdBat + nn]);
      D[(long)mm * ldD + nn] = vv;
    }
  }
}

// ---------------------------------------------------------------------------
// Naive fp32 GEMM for shapes WMMA can't tile (K=3/6, M=13, Ncols=4 FCs).
// ---------------------------------------------------------------------------
__global__ void k_gemm_naive(int M, int K, int Ncols,
                             const float* A, long Ar, long Ac, long Abat,
                             const float* B, long Br, long Bc, long Bbat,
                             float* D, long ldD, long Dbat,
                             const float* scale, const float* shift, int act) {
  long idx = (long)blockIdx.x * blockDim.x + threadIdx.x;
  long total = (long)M * Ncols;
  if (idx >= total) return;
  int b = blockIdx.y;
  int m = (int)(idx / Ncols);
  int n = (int)(idx % Ncols);
  const float* Ab = A + (long)b * Abat;
  const float* Bb = B + (long)b * Bbat;
  float acc = 0.f;
  for (int k = 0; k < K; ++k)
    acc += Ab[(long)m * Ar + (long)k * Ac] * Bb[(long)k * Br + (long)n * Bc];
  if (scale) acc = acc * scale[m] + (shift ? shift[m] : 0.f);
  else if (shift) acc += shift[m];
  if (act == 1)      acc = acc > 0.f ? acc : 0.f;
  else if (act == 2) acc = acc > 0.f ? acc : 0.2f * acc;
  D[(long)b * Dbat + (long)m * ldD + n] = acc;
}

// x' = einsum('bcn,bcd->bdn', x, T)
__global__ void k_transform(const float* x, const float* T, float* xT) {
  int idx = blockIdx.x * 256 + threadIdx.x;
  if (idx >= BATCH * 3 * NPTS) return;
  int n = idx % NPTS;
  int d = (idx / NPTS) % 3;
  int b = idx / (3 * NPTS);
  float s = 0.f;
  for (int c = 0; c < 3; ++c)
    s += x[((long)b * 3 + c) * NPTS + n] * T[b * 9 + c * 3 + d];
  xT[idx] = s;
}

// xx[b][n] = sum_c x[b][c][n]^2
__global__ void k_xx(const float* x, int C, float* xx) {
  int idx = blockIdx.x * 256 + threadIdx.x;
  if (idx >= BATCH * NPTS) return;
  int n = idx % NPTS, b = idx / NPTS;
  float s = 0.f;
  for (int c = 0; c < C; ++c) {
    float v = x[((long)b * C + c) * NPTS + n];
    s += v * v;
  }
  xx[idx] = s;
}

// Top-K=20 by key = 2*G[n][m] - xx[m] (row-constant -xx[n] doesn't affect
// ordering). gram==null -> compute 3-dim dot directly from x.
__global__ void k_knn(const float* x3, const float* gram, const float* xx,
                      int* idx_out) {
  int row = blockIdx.x * blockDim.x + threadIdx.x;
  if (row >= BATCH * NPTS) return;
  int b = row / NPTS, n = row % NPTS;
  float vals[KNN];
  int   idxs[KNN];
#pragma unroll
  for (int j = 0; j < KNN; ++j) { vals[j] = -3.0e38f; idxs[j] = 0; }
  const float* xxb = xx + (long)b * NPTS;
  const float* gr = gram ? gram + ((long)b * NPTS + n) * NPTS : nullptr;
  float xn0 = 0.f, xn1 = 0.f, xn2 = 0.f;
  const float* xb = nullptr;
  if (!gr) {
    xb = x3 + (long)b * 3 * NPTS;
    xn0 = xb[n]; xn1 = xb[NPTS + n]; xn2 = xb[2 * NPTS + n];
  }
  for (int m = 0; m < NPTS; ++m) {
    float g = gr ? gr[m] : (xn0 * xb[m] + xn1 * xb[NPTS + m] + xn2 * xb[2 * NPTS + m]);
    float key = 2.f * g - xxb[m];
    int cand = m;
#pragma unroll
    for (int j = 0; j < KNN; ++j) {
      if (key > vals[j]) {
        float tv = vals[j]; vals[j] = key; key = tv;
        int ti = idxs[j]; idxs[j] = cand; cand = ti;
      }
    }
  }
  int* o = idx_out + ((long)b * NPTS + n) * KNN;
#pragma unroll
  for (int j = 0; j < KNN; ++j) o[j] = idxs[j];
}

// edge features: out[b][c][n][k] = x[c][idx]-x[c][n] (c<C) ; x[c][n] (c>=C)
__global__ void k_gather(const float* x, int C, const int* idx, float* out) {
  long total = (long)BATCH * 2 * C * NPTS * KNN;
  long i = (long)blockIdx.x * 256 + threadIdx.x;
  if (i >= total) return;
  int k = (int)(i % KNN);
  long t = i / KNN;
  int n = (int)(t % NPTS); t /= NPTS;
  int c2 = (int)(t % (2 * C));
  int b = (int)(t / (2 * C));
  int c = c2 % C;
  float ctr = x[((long)b * C + c) * NPTS + n];
  float v;
  if (c2 < C) {
    int j = idx[((long)b * NPTS + n) * KNN + k];
    v = x[((long)b * C + c) * NPTS + j] - ctr;
  } else {
    v = ctr;
  }
  out[i] = v;
}

__global__ void k_maxk(const float* in, int C, float* out) {
  long i = (long)blockIdx.x * 256 + threadIdx.x;
  if (i >= (long)BATCH * C * NPTS) return;
  const float* p = in + i * KNN;
  float m = p[0];
  for (int k = 1; k < KNN; ++k) m = fmaxf(m, p[k]);
  out[i] = m;
}

__global__ void k_maxn(const float* in, int C, float* out) {
  int i = blockIdx.x * 256 + threadIdx.x;
  if (i >= BATCH * C) return;
  const float* p = in + (long)i * NPTS;
  float m = p[0];
  for (int n = 1; n < NPTS; ++n) m = fmaxf(m, p[n]);
  out[i] = m;
}

// in-place row softmax over last dim (row length NPTS); one block per row
__global__ __launch_bounds__(256) void k_softmax(float* a) {
  __shared__ float red[256];
  float* p = a + (long)blockIdx.x * NPTS;
  int tid = threadIdx.x;
  float mx = -3.0e38f;
  for (int i = tid; i < NPTS; i += 256) mx = fmaxf(mx, p[i]);
  red[tid] = mx;
  __syncthreads();
  for (int s = 128; s > 0; s >>= 1) {
    if (tid < s) red[tid] = fmaxf(red[tid], red[tid + s]);
    __syncthreads();
  }
  mx = red[0];
  __syncthreads();
  float sum = 0.f;
  for (int i = tid; i < NPTS; i += 256) {
    float e = __expf(p[i] - mx);
    p[i] = e;
    sum += e;
  }
  red[tid] = sum;
  __syncthreads();
  for (int s = 128; s > 0; s >>= 1) {
    if (tid < s) red[tid] += red[tid + s];
    __syncthreads();
  }
  float inv = 1.f / red[0];
  for (int i = tid; i < NPTS; i += 256) p[i] *= inv;
}

// cs[b][m] = sum_n attn[b][n][m]
__global__ void k_colsum(const float* a, float* cs) {
  int i = blockIdx.x * 256 + threadIdx.x;
  if (i >= BATCH * NPTS) return;
  int b = i / NPTS, m = i % NPTS;
  const float* p = a + (long)b * NPTS * NPTS + m;
  float s = 0.f;
  for (int n = 0; n < NPTS; ++n) s += p[(long)n * NPTS];
  cs[i] = s;
}

__global__ void k_sub(const float* a, const float* b, float* o, long n) {
  long i = (long)blockIdx.x * 256 + threadIdx.x;
  if (i < n) o[i] = a[i] - b[i];
}
__global__ void k_add(const float* a, const float* b, float* o, long n) {
  long i = (long)blockIdx.x * 256 + threadIdx.x;
  if (i < n) o[i] = a[i] + b[i];
}

// out[b][coff+c][n] = in[b][c][n]  (VALU fallback path)
__global__ void k_copych(const float* in, int Cin, float* out, int Cout, int coff) {
  long i = (long)blockIdx.x * 256 + threadIdx.x;
  if (i >= (long)BATCH * Cin * NPTS) return;
  int n = (int)(i % NPTS);
  long t = i / NPTS;
  int c = (int)(t % Cin);
  int b = (int)(t / Cin);
  out[((long)b * Cout + coff + c) * NPTS + n] = in[i];
}

#if HAVE_TDM
// ---------------------------------------------------------------------------
// Channel-concat copy via the Tensor Data Mover: each 1-wave workgroup DMAs
// an 8-row x 2048-col fp32 tile global->LDS, waits TENSORcnt, then DMAs
// LDS->global at the channel-offset destination. Descriptors per
// CDNA5 ISA: group0 = {count=1 | lds_addr | global_addr |
// type=2}, group1 = {data_size=4B, tensor_dim0/1, tile_dim0/1, dim0 stride}.
// 6-arg builtin form (clang-23): (v4u g0, v8i g1, v4i, v4i, v8i, i32 cpol).
// ---------------------------------------------------------------------------
#define TDM_CH 8
__global__ __launch_bounds__(32)
void k_copych_tdm(const float* in, int Cin, float* out, int Cout, int coff) {
  __shared__ float tile[TDM_CH * NPTS];    // 64KB
  int nch = Cin / TDM_CH;
  int b  = blockIdx.x / nch;
  int c0 = (blockIdx.x % nch) * TDM_CH;

  unsigned long long src =
      (unsigned long long)(const void*)(in + ((long)b * Cin + c0) * NPTS);
  unsigned long long dst =
      (unsigned long long)(void*)(out + ((long)b * Cout + coff + c0) * NPTS);
  unsigned int lds = (unsigned int)(unsigned long long)(void*)&tile[0];

  // D# group1 (256b): workgroup_mask=0, data_size=2 (4B), no pad/iterate;
  // tensor_dim0 = NPTS (bits 79:48), tensor_dim1 = TDM_CH (bits 111:80),
  // tile_dim0 = NPTS (bits 127:112), tile_dim1 = TDM_CH (bits 143:128),
  // tensor_dim0_stride = NPTS (bits 207:160).
  v8i g1;
  g1[0] = (int)(2u << 16);
  g1[1] = (int)((NPTS & 0xFFFFu) << 16);
  g1[2] = (int)((unsigned)TDM_CH << 16);
  g1[3] = (int)((NPTS & 0xFFFFu) << 16);
  g1[4] = TDM_CH;
  g1[5] = NPTS;
  g1[6] = 0;
  g1[7] = 0;
  v4i z4 = {0, 0, 0, 0};
  v8i z8 = {0, 0, 0, 0, 0, 0, 0, 0};

  v4u g0;
  g0[0] = 1u;                                    // count=1, user mode
  g0[1] = lds;                                   // lds_addr (bytes)
  g0[2] = (unsigned int)(src & 0xFFFFFFFFu);     // global_addr[31:0]
  g0[3] = (unsigned int)((src >> 32) & 0x01FFFFFFu) | (2u << 30);  // type=2
  __builtin_amdgcn_tensor_load_to_lds(g0, g1, z4, z4, z8, 0);
  __builtin_amdgcn_s_wait_tensorcnt(0);

  g0[2] = (unsigned int)(dst & 0xFFFFFFFFu);
  g0[3] = (unsigned int)((dst >> 32) & 0x01FFFFFFu) | (2u << 30);
  __builtin_amdgcn_tensor_store_from_lds(g0, g1, z4, z4, z8, 0);
  __builtin_amdgcn_s_wait_tensorcnt(0);
}
#endif

// out[b][coff+c][n] = gmax[b][c]  (broadcast global feature along N)
__global__ void k_bcast(const float* gmax, int Cb, float* out, int Cout, int coff) {
  long i = (long)blockIdx.x * 256 + threadIdx.x;
  if (i >= (long)BATCH * Cb * NPTS) return;
  int n = (int)(i % NPTS);
  long t = i / NPTS;
  int c = (int)(t % Cb);
  int b = (int)(t / Cb);
  out[((long)b * Cout + coff + c) * NPTS + n] = gmax[b * Cb + c];
}

// T[b] = fc3out.reshape(3,3) + I ; fc3out stored [9 rows][4 batch cols]
__global__ void k_buildT(const float* fc3, float* Tws, float* Tout) {
  int i = threadIdx.x;
  if (i >= 36) return;
  int b = i / 9, r = i % 9;
  int ii = r / 3, jj = r % 3;
  float v = fc3[r * 4 + b] + (ii == jj ? 1.f : 0.f);
  Tws[i] = v;
  Tout[i] = v;
}

// ---------------------------------------------------------------------------
// Host-side pipeline
// ---------------------------------------------------------------------------

// d_in index map: x, alignment scalar, then params leaves in sorted-key
// pytree order (bn tuples expand to gamma,beta,mean,var).
enum {
  IN_X = 0,
  P_BN1 = 2, P_BN2 = 6, P_BN3 = 10, P_BN4 = 14, P_BN5 = 18, P_BN6 = 22, P_BNG = 26,
  P_C1W = 30, P_C2W = 31, P_C3W = 32, P_C4W = 33,
  P_C5B = 34, P_C5W = 35, P_C6B = 36, P_C6W = 37, P_C7B = 38, P_C7W = 39,
  P_CGW = 40,
  P_SA1 = 41, P_SA2 = 51, P_SA3 = 61, P_SA4 = 71,  // {bn(4), k, q, t_b, t_w, v_b, v_w}
  P_STN = 81
};
enum {  // stn sub-offsets (sorted keys): bn1..bn5, c1_b,c1_w,...,fc3_b,fc3_w
  S_BN1 = 0, S_BN2 = 4, S_BN3 = 8, S_BN4 = 12, S_BN5 = 16,
  S_C1B = 20, S_C1W = 21, S_C2B = 22, S_C2W = 23, S_C3B = 24, S_C3W = 25,
  S_FC1B = 26, S_FC1W = 27, S_FC2B = 28, S_FC2W = 29, S_FC3B = 30, S_FC3W = 31
};
enum { A_K = 4, A_Q = 5, A_TB = 6, A_TW = 7, A_VB = 8, A_VW = 9 };  // sa sub

extern "C" void kernel_launch(void* const* d_in, const int* in_sizes, int n_in,
                              void* d_out, int out_size, void* d_ws, size_t ws_size,
                              hipStream_t stream) {
  (void)in_sizes; (void)n_in; (void)out_size; (void)ws_size;
  const float* X = (const float*)d_in[IN_X];
  float* OUT = (float*)d_out;
  float* SEG = OUT;                                 // [B,13,N]
  float* TOUT = OUT + (long)BATCH * NSEG * NPTS;    // [B,3,3]

  auto in = [&](int i) { return (const float*)d_in[i]; };

  // ---- workspace layout (floats), deterministic bump allocator ------------
  float* Wf = (float*)d_ws;
  size_t cur = 0;
  auto alloc = [&](size_t nfloats) {
    float* r = Wf + cur;
    cur += (nfloats + 63) & ~(size_t)63;
    return r;
  };
  float* SC   = alloc(1024);
  float* SH   = alloc(1024);
  float* TWS  = alloc(64);
  float* XXB  = alloc(BATCH * NPTS);
  float* POOL = alloc(BATCH * 1024);
  float* FC1O = alloc(512 * 4);
  float* FC2O = alloc(256 * 4);
  float* FC3O = alloc(64);
  float* CS   = alloc(BATCH * NPTS);
  float* GMX  = alloc(BATCH * 1024);
  float* XT   = alloc((size_t)BATCH * 3 * NPTS);
  int*   IDX  = (int*)alloc((size_t)BATCH * NPTS * KNN);
  float* X1   = alloc((size_t)BATCH * 64 * NPTS);
  float* X2   = alloc((size_t)BATCH * 64 * NPTS);
  float* H0   = alloc((size_t)BATCH * 128 * NPTS);
  float* H1   = alloc((size_t)BATCH * 128 * NPTS);
  float* H2   = alloc((size_t)BATCH * 128 * NPTS);
  float* H3   = alloc((size_t)BATCH * 128 * NPTS);
  float* H4   = alloc((size_t)BATCH * 128 * NPTS);
  float* QB   = alloc((size_t)BATCH * 32 * NPTS);
  float* KB   = alloc((size_t)BATCH * 32 * NPTS);
  float* VB   = alloc((size_t)BATCH * 128 * NPTS);
  float* XR   = alloc((size_t)BATCH * 128 * NPTS);
  float* DT   = alloc((size_t)BATCH * 128 * NPTS);
  float* TRB  = alloc((size_t)BATCH * 128 * NPTS);
  float* NNB  = alloc((size_t)BATCH * NPTS * NPTS);       // 16.8M: gram + attn
  float* U    = alloc((size_t)41943040);                  // phase-reused union
  // union overlays (lifetimes disjoint by phase):
  float* STNH3 = NNB;                       // [B,1024,N] (dead before gram)
  float* STNH1 = U;                         // [B,64,N]
  float* STNH2 = U + 1048576;               // [B,128,N]
  float* EDGE  = U;                         // [B,<=128,N,K]
  float* EO1   = U + 20971520;              // [B,64,N,K]
  float* EO2   = U + 31457280;              // [B,64,N,K]
  float* CAT5  = U;                         // [B,512,N]
  float* GBUF  = U + 4194304;               // [B,1024,N]
  float* CAT15 = U + 12582912;              // [B,1536,N]
  float* C5    = U + 25165824;              // [B,512,N]
  float* C6    = U + 29360128;              // [B,256,N]

  // ---- launch helpers -----------------------------------------------------
  auto fold = [&](int base, const float* bias, int C) {
    k_fold_bn<<<(C + 255) / 256, 256, 0, stream>>>(
        in(base), in(base + 1), in(base + 2), in(base + 3), bias, SC, SH, C);
  };
  auto gemmW = [&](int M, int K, int Nc,
                   const float* A, long Ar, long Ac, long Abat,
                   const float* B, long Br, long Bc, long Bbat,
                   float* D, long ldD, long Dbat,
                   const float* sc, const float* sh,
                   const float* cd, long cdBat, int act, int nb) {
    dim3 g(Nc / 128, M / 32, nb);
    k_gemm_wmma<<<g, 256, 0, stream>>>(M, K, Nc, A, Ar, Ac, Abat,
                                       B, Br, Bc, Bbat, D, ldD, Dbat,
                                       sc, sh, cd, cdBat, act);
  };
  auto gemmN = [&](int M, int K, int Nc,
                   const float* A, long Ar, long Ac, long Abat,
                   const float* B, long Br, long Bc, long Bbat,
                   float* D, long ldD, long Dbat,
                   const float* sc, const float* sh, int act, int nb) {
    long total = (long)M * Nc;
    dim3 g((unsigned)((total + 255) / 256), nb);
    k_gemm_naive<<<g, 256, 0, stream>>>(M, K, Nc, A, Ar, Ac, Abat,
                                        B, Br, Bc, Bbat, D, ldD, Dbat,
                                        sc, sh, act);
  };
  auto gblk = [](long n) { return (unsigned)((n + 255) / 256); };
  auto copych = [&](const float* src, int Cin, float* dst, int Cout, int coff) {
#if HAVE_TDM
    k_copych_tdm<<<BATCH * (Cin / TDM_CH), 32, 0, stream>>>(src, Cin, dst, Cout, coff);
#else
    k_copych<<<gblk((long)BATCH * Cin * NPTS), 256, 0, stream>>>(src, Cin, dst, Cout, coff);
#endif
  };

  const long NK = (long)NPTS * KNN;          // 40960 columns for edge convs
  const long SB3 = 3L * NPTS, SB64 = 64L * NPTS, SB128 = 128L * NPTS;

  // ========================== STN =========================================
  fold(P_STN + S_BN1, in(P_STN + S_C1B), 64);
  gemmN(64, 3, NPTS, in(P_STN + S_C1W), 3, 1, 0, X, NPTS, 1, SB3,
        STNH1, NPTS, SB64, SC, SH, 1, BATCH);
  fold(P_STN + S_BN2, in(P_STN + S_C2B), 128);
  gemmW(128, 64, NPTS, in(P_STN + S_C2W), 64, 1, 0, STNH1, NPTS, 1, SB64,
        STNH2, NPTS, SB128, SC, SH, nullptr, 0, 1, BATCH);
  fold(P_STN + S_BN3, in(P_STN + S_C3B), 1024);
  gemmW(1024, 128, NPTS, in(P_STN + S_C3W), 128, 1, 0, STNH2, NPTS, 1, SB128,
        STNH3, NPTS, 1024L * NPTS, SC, SH, nullptr, 0, 1, BATCH);
  k_maxn<<<(BATCH * 1024 + 255) / 256, 256, 0, stream>>>(STNH3, 1024, POOL);
  fold(P_STN + S_BN4, in(P_STN + S_FC1B), 512);
  gemmN(512, 1024, BATCH, in(P_STN + S_FC1W), 1024, 1, 0, POOL, 1, 1024, 0,
        FC1O, BATCH, 0, SC, SH, 1, 1);
  fold(P_STN + S_BN5, in(P_STN + S_FC2B), 256);
  gemmN(256, 512, BATCH, in(P_STN + S_FC2W), 512, 1, 0, FC1O, BATCH, 1, 0,
        FC2O, BATCH, 0, SC, SH, 1, 1);
  gemmN(9, 256, BATCH, in(P_STN + S_FC3W), 256, 1, 0, FC2O, BATCH, 1, 0,
        FC3O, BATCH, 0, nullptr, in(P_STN + S_FC3B), 0, 1);
  k_buildT<<<1, 64, 0, stream>>>(FC3O, TWS, TOUT);
  k_transform<<<gblk(BATCH * 3 * NPTS), 256, 0, stream>>>(X, TWS, XT);

  // ===================== EdgeConv stage 1 (C=3) ===========================
  k_xx<<<gblk(BATCH * NPTS), 256, 0, stream>>>(XT, 3, XXB);
  k_knn<<<gblk(BATCH * NPTS), 256, 0, stream>>>(XT, nullptr, XXB, IDX);
  k_gather<<<gblk((long)BATCH * 6 * NK), 256, 0, stream>>>(XT, 3, IDX, EDGE);
  fold(P_BN1, nullptr, 64);
  gemmN(64, 6, (int)NK, in(P_C1W), 6, 1, 0, EDGE, NK, 1, 6 * NK,
        EO1, NK, 64 * NK, SC, SH, 2, BATCH);
  fold(P_BN2, nullptr, 64);
  gemmW(64, 64, (int)NK, in(P_C2W), 64, 1, 0, EO1, NK, 1, 64 * NK,
        EO2, NK, 64 * NK, SC, SH, nullptr, 0, 2, BATCH);
  k_maxk<<<gblk((long)BATCH * 64 * NPTS), 256, 0, stream>>>(EO2, 64, X1);

  // ===================== EdgeConv stage 2 (C=64) ==========================
  k_xx<<<gblk(BATCH * NPTS), 256, 0, stream>>>(X1, 64, XXB);
  // Gram = x1^T x1 via WMMA (A accessed transposed through strides)
  gemmW(NPTS, 64, NPTS, X1, 1, NPTS, SB64, X1, NPTS, 1, SB64,
        NNB, NPTS, (long)NPTS * NPTS, nullptr, nullptr, nullptr, 0, 0, BATCH);
  k_knn<<<gblk(BATCH * NPTS), 256, 0, stream>>>(nullptr, NNB, XXB, IDX);
  k_gather<<<gblk((long)BATCH * 128 * NK), 256, 0, stream>>>(X1, 64, IDX, EDGE);
  fold(P_BN3, nullptr, 64);
  gemmW(64, 128, (int)NK, in(P_C3W), 128, 1, 0, EDGE, NK, 1, 128 * NK,
        EO1, NK, 64 * NK, SC, SH, nullptr, 0, 2, BATCH);
  fold(P_BN4, nullptr, 64);
  gemmW(64, 64, (int)NK, in(P_C4W), 64, 1, 0, EO1, NK, 1, 64 * NK,
        EO2, NK, 64 * NK, SC, SH, nullptr, 0, 2, BATCH);
  k_maxk<<<gblk((long)BATCH * 64 * NPTS), 256, 0, stream>>>(EO2, 64, X2);

  // h0 = concat(x1, x2) along channels -> [B,128,N]  (TDM block copies)
  copych(X1, 64, H0, 128, 0);
  copych(X2, 64, H0, 128, 64);

  // ===================== 4 offset self-attention blocks ===================
  const float* hin = H0;
  float* houts[4] = {H1, H2, H3, H4};
  int bases[4] = {P_SA1, P_SA2, P_SA3, P_SA4};
  for (int s = 0; s < 4; ++s) {
    int pb = bases[s];
    float* hout = houts[s];
    // q,k projections [B,32,N]; v projection [B,128,N] (+bias)
    gemmW(32, 128, NPTS, in(pb + A_Q), 128, 1, 0, hin, NPTS, 1, SB128,
          QB, NPTS, 32L * NPTS, nullptr, nullptr, nullptr, 0, 0, BATCH);
    gemmW(32, 128, NPTS, in(pb + A_K), 128, 1, 0, hin, NPTS, 1, SB128,
          KB, NPTS, 32L * NPTS, nullptr, nullptr, nullptr, 0, 0, BATCH);
    gemmW(128, 128, NPTS, in(pb + A_VW), 128, 1, 0, hin, NPTS, 1, SB128,
          VB, NPTS, SB128, nullptr, in(pb + A_VB), nullptr, 0, 0, BATCH);
    // scores[n][m] = sum_o q[o][n] * k[o][m]  (q transposed via strides)
    gemmW(NPTS, 32, NPTS, QB, 1, NPTS, 32L * NPTS, KB, NPTS, 1, 32L * NPTS,
          NNB, NPTS, (long)NPTS * NPTS, nullptr, nullptr, nullptr, 0, 0, BATCH);
    k_softmax<<<BATCH * NPTS, 256, 0, stream>>>(NNB);
    k_colsum<<<gblk(BATCH * NPTS), 256, 0, stream>>>(NNB, CS);
    // xr = (v . attn) with column re-normalization fused in the epilogue
    gemmW(128, NPTS, NPTS, VB, NPTS, 1, SB128, NNB, NPTS, 1, (long)NPTS * NPTS,
          XR, NPTS, SB128, nullptr, nullptr, CS, NPTS, 0, BATCH);
    k_sub<<<gblk((long)BATCH * 128 * NPTS), 256, 0, stream>>>(
        hin, XR, DT, (long)BATCH * 128 * NPTS);
    fold(pb, in(pb + A_TB), 128);  // sa bn at pb+0..3
    gemmW(128, 128, NPTS, in(pb + A_TW), 128, 1, 0, DT, NPTS, 1, SB128,
          TRB, NPTS, SB128, SC, SH, nullptr, 0, 1, BATCH);
    k_add<<<gblk((long)BATCH * 128 * NPTS), 256, 0, stream>>>(
        hin, TRB, hout, (long)BATCH * 128 * NPTS);
    hin = hout;
  }

  // ===================== head =============================================
  for (int s = 0; s < 4; ++s)
    copych(houts[s], 128, CAT5, 512, 128 * s);
  fold(P_BNG, nullptr, 1024);
  gemmW(1024, 512, NPTS, in(P_CGW), 512, 1, 0, CAT5, NPTS, 1, 512L * NPTS,
        GBUF, NPTS, 1024L * NPTS, SC, SH, nullptr, 0, 2, BATCH);
  k_maxn<<<(BATCH * 1024 + 255) / 256, 256, 0, stream>>>(GBUF, 1024, GMX);
  k_bcast<<<gblk((long)BATCH * 1024 * NPTS), 256, 0, stream>>>(GMX, 1024, CAT15, 1536, 0);
  copych(CAT5, 512, CAT15, 1536, 1024);
  fold(P_BN5, in(P_C5B), 512);
  gemmW(512, 1536, NPTS, in(P_C5W), 1536, 1, 0, CAT15, NPTS, 1, 1536L * NPTS,
        C5, NPTS, 512L * NPTS, SC, SH, nullptr, 0, 1, BATCH);
  fold(P_BN6, in(P_C6B), 256);
  gemmW(256, 512, NPTS, in(P_C6W), 512, 1, 0, C5, NPTS, 1, 512L * NPTS,
        C6, NPTS, 256L * NPTS, SC, SH, nullptr, 0, 1, BATCH);
  gemmN(NSEG, 256, NPTS, in(P_C7W), 256, 1, 0, C6, NPTS, 1, 256L * NPTS,
        SEG, NPTS, (long)NSEG * NPTS, nullptr, in(P_C7B), 0, BATCH);
}
